// DecoderWithAttention_21861383537140
// MI455X (gfx1250) — compile-verified, matrix operand order checked by reading, not run on hardware
//
#include <hip/hip_runtime.h>

typedef __bf16 bf16_t;
typedef __attribute__((ext_vector_type(16))) __bf16 v16bf;
typedef __attribute__((ext_vector_type(8)))  __bf16 v8bf;
typedef __attribute__((ext_vector_type(8)))  float  v8f;

#define B_     64
#define NPIX   196
#define ENC_   2048
#define DEC_   512
#define ATT_   512
#define EMB_   512
#define VOCAB_ 32000
#define L_     22
#define T_     21
#define XDIM   3072   // [x_emb(512) | z(2048) | h(512)]

// ---------------------------------------------------------------------------
// WMMA fragment load: 16-bit A/B 16x32 tile, row-major source with row stride K.
// Lane l<16 holds row l, K = {khalf..khalf+7} u {khalf+16..khalf+23} (khalf=0);
// lane l>=16 holds row l-16 with khalf=8. Two 16B chunks -> v16bf.
// ---------------------------------------------------------------------------
__device__ __forceinline__ v16bf load_frag(const bf16_t* p) {
    v8bf lo = *(const v8bf*)(p);
    v8bf hi = *(const v8bf*)(p + 16);
    return __builtin_shufflevector(lo, hi, 0,1,2,3,4,5,6,7,8,9,10,11,12,13,14,15);
}

// Generic bf16 WMMA GEMM: C(MxN,fp32,ldc) = A(MxK) * Bt(NxK)^T + bias.
// Optional per-row mask: if mask_lens && t >= mask_lens[m], output 0.
// Block = 256 threads = 8 waves. Each wave computes a 16x64 strip (4 N-tiles),
// reusing the A fragment across 4 WMMAs per K-step (5 loads / 4 wmma).
// Block tile = 32 (M) x 256 (N). Requires M%32==0, N%256==0, K%32==0.
__global__ void __launch_bounds__(256)
gemm_bf16_wmma(const bf16_t* __restrict__ A, const bf16_t* __restrict__ Bt,
               const float* __restrict__ bias, float* __restrict__ C,
               int M, int N, int K, int ldc,
               const int* __restrict__ mask_lens, int t)
{
    const int wave = threadIdx.x >> 5;
    const int lane = threadIdx.x & 31;
    const int m0 = (blockIdx.y * 2 + (wave >> 2)) * 16;
    const int n0 = (blockIdx.x * 4 + (wave & 3)) * 64;   // 4 consecutive 16-col tiles
    if (m0 >= M || n0 >= N) return;

    const int r     = lane & 15;
    const int khalf = (lane >> 4) << 3;       // 0 or 8
    const bf16_t* ap = A  + (size_t)(m0 + r) * K + khalf;
    const bf16_t* bp = Bt + (size_t)(n0 + r) * K + khalf;
    const size_t  bstep = (size_t)16 * K;     // next 16-col tile of Bt

    v8f acc0 = {}, acc1 = {}, acc2 = {}, acc3 = {};
    for (int k = 0; k < K; k += 32) {
        __builtin_prefetch(ap + k + 256, 0, 1);
        __builtin_prefetch(bp + k + 256, 0, 1);
        const v16bf af = load_frag(ap + k);
        const v16bf b0 = load_frag(bp + k);
        const v16bf b1 = load_frag(bp + bstep + k);
        const v16bf b2 = load_frag(bp + 2 * bstep + k);
        const v16bf b3 = load_frag(bp + 3 * bstep + k);
        acc0 = __builtin_amdgcn_wmma_f32_16x16x32_bf16(false, af, false, b0, (short)0, acc0, false, false);
        acc1 = __builtin_amdgcn_wmma_f32_16x16x32_bf16(false, af, false, b1, (short)0, acc1, false, false);
        acc2 = __builtin_amdgcn_wmma_f32_16x16x32_bf16(false, af, false, b2, (short)0, acc2, false, false);
        acc3 = __builtin_amdgcn_wmma_f32_16x16x32_bf16(false, af, false, b3, (short)0, acc3, false, false);
    }

    // D layout: VGPR i -> M = m0 + i (lanes 0-15) / m0 + 8 + i (lanes 16-31)
    const int nl = lane & 15;
    const int mb = m0 + ((lane >> 4) << 3);
    v8f accs[4] = {acc0, acc1, acc2, acc3};
#pragma unroll
    for (int j = 0; j < 4; ++j) {
        const int n = n0 + j * 16 + nl;
        const float bv = bias ? bias[n] : 0.0f;
#pragma unroll
        for (int i = 0; i < 8; ++i) {
            const int m = mb + i;
            float v = accs[j][i] + bv;
            if (mask_lens && t >= mask_lens[m]) v = 0.0f;
            C[(size_t)m * ldc + n] = v;
        }
    }
}

// ---------------------------------------------------------------------------
// Stable descending argsort of lens (64 entries) + caps gather + int outputs.
// ---------------------------------------------------------------------------
__global__ void sort_kernel(const int* __restrict__ lens2d,
                            const int* __restrict__ caps,
                            float* __restrict__ out_caps,
                            float* __restrict__ out_dlen,
                            float* __restrict__ out_sind,
                            int* __restrict__ ws_sind,
                            int* __restrict__ ws_dlen)
{
    const int i = threadIdx.x;          // 0..63
    __shared__ int lens[B_];
    __shared__ int sind[B_];
    __shared__ int dlen[B_];
    const int li = lens2d[i];           // caption_lengths[:,0]
    lens[i] = li;
    __syncthreads();
    int rank = 0;
    for (int j = 0; j < B_; ++j) {
        const int lj = lens[j];
        if (lj > li || (lj == li && j < i)) rank++;
    }
    sind[rank] = i;
    dlen[rank] = li - 1;
    __syncthreads();
    ws_sind[i] = sind[i];
    ws_dlen[i] = dlen[i];
    out_dlen[i] = (float)dlen[i];
    out_sind[i] = (float)sind[i];
    for (int k = 0; k < L_; ++k)
        out_caps[i * L_ + k] = (float)caps[sind[i] * L_ + k];
}

__global__ void zero_f32(float* __restrict__ p, int n) {
    int i = blockIdx.x * blockDim.x + threadIdx.x;
    if (i < n) p[i] = 0.0f;
}

__global__ void f32_to_bf16(const float* __restrict__ in, bf16_t* __restrict__ out, int n) {
    int i = blockIdx.x * blockDim.x + threadIdx.x;
    if (i < n) out[i] = (bf16_t)in[i];
}

// W (K x N) fp32 -> Wt (N x K) bf16
__global__ void transpose_to_bf16(const float* __restrict__ W, bf16_t* __restrict__ Wt,
                                  int K, int N)
{
    const size_t total  = (size_t)K * N;
    const size_t stride = (size_t)gridDim.x * blockDim.x;
    for (size_t i = (size_t)blockIdx.x * blockDim.x + threadIdx.x; i < total; i += stride) {
        const size_t k = i / N, n = i % N;
        Wt[n * (size_t)K + k] = (bf16_t)W[i];
    }
}

// Combined LSTM weight: Wt (2048 x 3072) where rows of [W_ih(2560x2048); W_hh(512x2048)]
// become K; bias_comb = b_ih + b_hh.
__global__ void build_comb(const float* __restrict__ W_ih, const float* __restrict__ W_hh,
                           const float* __restrict__ b_ih, const float* __restrict__ b_hh,
                           bf16_t* __restrict__ Wt, float* __restrict__ bias)
{
    const size_t total  = (size_t)(4 * DEC_) * XDIM;
    const size_t stride = (size_t)gridDim.x * blockDim.x;
    const size_t gid = (size_t)blockIdx.x * blockDim.x + threadIdx.x;
    for (size_t i = gid; i < total; i += stride) {
        const size_t n = i / XDIM, k = i % XDIM;
        const float v = (k < (EMB_ + ENC_)) ? W_ih[k * (4 * DEC_) + n]
                                            : W_hh[(k - (EMB_ + ENC_)) * (4 * DEC_) + n];
        Wt[i] = (bf16_t)v;
    }
    if (gid < 4 * DEC_) bias[gid] = b_ih[gid] + b_hh[gid];
}

// enc (unsorted fp32) -> enc_s bf16 (sorted by sind)
__global__ void gather_enc_bf16(const float* __restrict__ enc, const int* __restrict__ sind,
                                bf16_t* __restrict__ out)
{
    const size_t per_b  = (size_t)NPIX * ENC_;
    const size_t total  = (size_t)B_ * per_b;
    const size_t stride = (size_t)gridDim.x * blockDim.x;
    for (size_t i = (size_t)blockIdx.x * blockDim.x + threadIdx.x; i < total; i += stride) {
        const size_t b = i / per_b, rem = i % per_b;
        out[i] = (bf16_t)enc[(size_t)sind[b] * per_b + rem];
    }
}

// mean over 196 pixels of sorted enc -> bf16 (64 x 2048)
__global__ void mean_enc_kernel(const float* __restrict__ enc, const int* __restrict__ sind,
                                bf16_t* __restrict__ mean_bf)
{
    const int b = blockIdx.x;
    const float* eb = enc + (size_t)sind[b] * NPIX * ENC_;
    for (int e = threadIdx.x; e < ENC_; e += blockDim.x) {
        float s = 0.0f;
        for (int p = 0; p < NPIX; ++p) s += eb[(size_t)p * ENC_ + e];
        mean_bf[(size_t)b * ENC_ + e] = (bf16_t)(s * (1.0f / NPIX));
    }
}

// ---------------------------------------------------------------------------
// Attention step: e = relu(pre_att + d_att)@v_att + bv; softmax over 196;
// z = alpha . enc_s; gate = h@W_beta + b_beta (no sigmoid, matches ref);
// build x = [emb[token] | gate*z | h] in bf16. One block per batch row.
// ---------------------------------------------------------------------------
__global__ void __launch_bounds__(256)
attention_step(const float* __restrict__ enc, const int* __restrict__ sind,
               const float* __restrict__ pre_att, const float* __restrict__ d_att,
               const float* __restrict__ v_att, const float* __restrict__ bv_att,
               const float* __restrict__ W_beta, const float* __restrict__ b_beta,
               const float* __restrict__ h, const float* __restrict__ emb,
               const int* __restrict__ caps, int t,
               bf16_t* __restrict__ x)
{
    const int b = blockIdx.x;
    const int tid = threadIdx.x;
    __shared__ float sh_e[NPIX];
    __shared__ float sh_alpha[NPIX];
    __shared__ float sh_red[256];
    __shared__ float sh_gate;
    __shared__ int   sh_tok;

    if (tid < NPIX) {
        const float* pa = pre_att + ((size_t)b * NPIX + tid) * ATT_;
        const float* da = d_att + (size_t)b * ATT_;
        float acc = 0.0f;
        for (int a = 0; a < ATT_; ++a) {
            float v = pa[a] + da[a];
            v = v > 0.0f ? v : 0.0f;
            acc += v * v_att[a];
        }
        sh_e[tid] = acc + bv_att[0];
    }
    {
        float acc = 0.0f;
        for (int j = tid; j < DEC_; j += 256) acc += h[(size_t)b * DEC_ + j] * W_beta[j];
        sh_red[tid] = acc;
    }
    __syncthreads();
    if (tid == 0) {
        float mx = -1e30f;
        for (int p = 0; p < NPIX; ++p) mx = fmaxf(mx, sh_e[p]);
        float s = 0.0f;
        for (int p = 0; p < NPIX; ++p) { float ev = __expf(sh_e[p] - mx); sh_alpha[p] = ev; s += ev; }
        const float inv = 1.0f / s;
        for (int p = 0; p < NPIX; ++p) sh_alpha[p] *= inv;
        float g = 0.0f;
        for (int j = 0; j < 256; ++j) g += sh_red[j];
        sh_gate = g + b_beta[0];
        sh_tok  = caps[sind[b] * L_ + t];
    }
    __syncthreads();

    const float gate = sh_gate;
    const int   sb   = sind[b];
    bf16_t* xr = x + (size_t)b * XDIM;

    for (int j = tid; j < EMB_; j += 256)
        xr[j] = (bf16_t)emb[(size_t)sh_tok * EMB_ + j];
    for (int j = tid; j < DEC_; j += 256)
        xr[EMB_ + ENC_ + j] = (bf16_t)h[(size_t)b * DEC_ + j];

    float acc[8];
#pragma unroll
    for (int i = 0; i < 8; ++i) acc[i] = 0.0f;
    const float* eb = enc + (size_t)sb * NPIX * ENC_;
    for (int p = 0; p < NPIX; ++p) {
        const float a = sh_alpha[p];
        const float* ep = eb + (size_t)p * ENC_ + tid;
#pragma unroll
        for (int i = 0; i < 8; ++i) acc[i] += a * ep[i * 256];
    }
#pragma unroll
    for (int i = 0; i < 8; ++i)
        xr[EMB_ + tid + i * 256] = (bf16_t)(gate * acc[i]);
}

// LSTM pointwise: gates(64x2048)->(i,f,g,o); update h,c with mask; emit bf16.
__global__ void lstm_update(const float* __restrict__ gates,
                            float* __restrict__ h, float* __restrict__ c,
                            bf16_t* __restrict__ h_bf, bf16_t* __restrict__ hnew_bf,
                            const int* __restrict__ dec_lens, int t)
{
    const int idx = blockIdx.x * blockDim.x + threadIdx.x;
    if (idx >= B_ * DEC_) return;
    const int b = idx / DEC_, j = idx % DEC_;
    const float* g = gates + (size_t)b * (4 * DEC_);
    const float ig = 1.0f / (1.0f + __expf(-g[j]));
    const float fg = 1.0f / (1.0f + __expf(-g[DEC_ + j]));
    const float gg = tanhf(g[2 * DEC_ + j]);
    const float og = 1.0f / (1.0f + __expf(-g[3 * DEC_ + j]));
    const float cn = fg * c[idx] + ig * gg;
    const float hn = og * tanhf(cn);
    const bool  m  = t < dec_lens[b];
    const float ho = m ? hn : h[idx];
    const float co = m ? cn : c[idx];
    h[idx] = ho; c[idx] = co;
    h_bf[idx]   = (bf16_t)ho;
    hnew_bf[idx] = (bf16_t)hn;
}

// ---------------------------------------------------------------------------
extern "C" void kernel_launch(void* const* d_in, const int* in_sizes, int n_in,
                              void* d_out, int out_size, void* d_ws, size_t ws_size,
                              hipStream_t stream)
{
    const float* enc      = (const float*)d_in[0];
    const int*   caps     = (const int*)  d_in[1];
    const int*   lens     = (const int*)  d_in[2];
    const float* emb      = (const float*)d_in[3];
    const float* We_att   = (const float*)d_in[4];
    const float* be_att   = (const float*)d_in[5];
    const float* Wd_att   = (const float*)d_in[6];
    const float* bd_att   = (const float*)d_in[7];
    const float* v_att    = (const float*)d_in[8];
    const float* bv_att   = (const float*)d_in[9];
    const float* W_beta   = (const float*)d_in[10];
    const float* b_beta   = (const float*)d_in[11];
    const float* W_ih     = (const float*)d_in[12];
    const float* b_ih     = (const float*)d_in[13];
    const float* W_hh     = (const float*)d_in[14];
    const float* b_hh     = (const float*)d_in[15];
    const float* W_init_h = (const float*)d_in[16];
    const float* b_init_h = (const float*)d_in[17];
    const float* W_init_c = (const float*)d_in[18];
    const float* b_init_c = (const float*)d_in[19];
    const float* W_fc     = (const float*)d_in[20];
    const float* b_fc     = (const float*)d_in[21];

    // ---- workspace bump allocator ----
    char* wsp = (char*)d_ws;
    auto alloc = [&](size_t bytes) -> char* {
        char* p = wsp;
        wsp += (bytes + 255) & ~(size_t)255;
        return p;
    };
    int*    sind      = (int*)   alloc(B_ * sizeof(int));
    int*    dlen      = (int*)   alloc(B_ * sizeof(int));
    bf16_t* enc_bf    = (bf16_t*)alloc((size_t)B_ * NPIX * ENC_ * 2);
    float*  pre_att   = (float*) alloc((size_t)B_ * NPIX * ATT_ * 4);
    bf16_t* WeT       = (bf16_t*)alloc((size_t)ATT_ * ENC_ * 2);
    bf16_t* WdT       = (bf16_t*)alloc((size_t)ATT_ * DEC_ * 2);
    bf16_t* WcombT    = (bf16_t*)alloc((size_t)(4 * DEC_) * XDIM * 2);
    bf16_t* WfcT      = (bf16_t*)alloc((size_t)VOCAB_ * DEC_ * 2);
    bf16_t* WinhT     = (bf16_t*)alloc((size_t)DEC_ * ENC_ * 2);
    bf16_t* WincT     = (bf16_t*)alloc((size_t)DEC_ * ENC_ * 2);
    float*  bias_comb = (float*) alloc((size_t)(4 * DEC_) * 4);
    bf16_t* mean_bf   = (bf16_t*)alloc((size_t)B_ * ENC_ * 2);
    float*  hbuf      = (float*) alloc((size_t)B_ * DEC_ * 4);
    float*  cbuf      = (float*) alloc((size_t)B_ * DEC_ * 4);
    bf16_t* h_bf      = (bf16_t*)alloc((size_t)B_ * DEC_ * 2);
    bf16_t* hnew_bf   = (bf16_t*)alloc((size_t)B_ * DEC_ * 2);
    float*  datt_buf  = (float*) alloc((size_t)B_ * ATT_ * 4);
    float*  gatesbuf  = (float*) alloc((size_t)B_ * 4 * DEC_ * 4);
    bf16_t* xbuf      = (bf16_t*)alloc((size_t)B_ * XDIM * 2);

    // ---- output layout (floats) ----
    float* out       = (float*)d_out;
    float* out_preds = out;                                    // 64*21*32000
    float* out_caps  = out_preds + (size_t)B_ * T_ * VOCAB_;   // 64*22
    float* out_dlen  = out_caps + B_ * L_;                     // 64
    float* out_alph  = out_dlen + B_;                          // 64*21*196
    float* out_sind  = out_alph + (size_t)B_ * T_ * NPIX;      // 64

    auto gemm = [&](const bf16_t* A, const bf16_t* Bt, const float* bias, float* C,
                    int M, int N, int K, int ldc, const int* mask, int t) {
        dim3 g(N / 256, M / 32);
        gemm_bf16_wmma<<<g, 256, 0, stream>>>(A, Bt, bias, C, M, N, K, ldc, mask, t);
    };

    // ---- setup ----
    sort_kernel<<<1, B_, 0, stream>>>(lens, caps, out_caps, out_dlen, out_sind, sind, dlen);
    {
        const int n = B_ * T_ * NPIX;
        zero_f32<<<(n + 255) / 256, 256, 0, stream>>>(out_alph, n);
    }
    const int G = 2048;
    transpose_to_bf16<<<G, 256, 0, stream>>>(We_att,   WeT,   ENC_, ATT_);
    transpose_to_bf16<<<G, 256, 0, stream>>>(Wd_att,   WdT,   DEC_, ATT_);
    transpose_to_bf16<<<G, 256, 0, stream>>>(W_fc,     WfcT,  DEC_, VOCAB_);
    transpose_to_bf16<<<G, 256, 0, stream>>>(W_init_h, WinhT, ENC_, DEC_);
    transpose_to_bf16<<<G, 256, 0, stream>>>(W_init_c, WincT, ENC_, DEC_);
    build_comb<<<G, 256, 0, stream>>>(W_ih, W_hh, b_ih, b_hh, WcombT, bias_comb);
    gather_enc_bf16<<<G, 256, 0, stream>>>(enc, sind, enc_bf);
    mean_enc_kernel<<<B_, 256, 0, stream>>>(enc, sind, mean_bf);

    // h0 = mean_enc @ W_init_h + b ; c0 likewise (WMMA)
    gemm(mean_bf, WinhT, b_init_h, hbuf, B_, DEC_, ENC_, DEC_, nullptr, 0);
    gemm(mean_bf, WincT, b_init_c, cbuf, B_, DEC_, ENC_, DEC_, nullptr, 0);
    f32_to_bf16<<<(B_ * DEC_ + 255) / 256, 256, 0, stream>>>(hbuf, h_bf, B_ * DEC_);

    // pre_att = enc_s @ We_att + be_att (12544 x 512, the big GEMM)
    gemm(enc_bf, WeT, be_att, pre_att, B_ * NPIX, ATT_, ENC_, ATT_, nullptr, 0);

    // ---- decode loop ----
    for (int t = 0; t < T_; ++t) {
        // d_att = h @ Wd_att + bd_att
        gemm(h_bf, WdT, bd_att, datt_buf, B_, ATT_, DEC_, ATT_, nullptr, 0);
        // attention + gate + build x
        attention_step<<<B_, 256, 0, stream>>>(enc, sind, pre_att, datt_buf,
                                               v_att, bv_att, W_beta, b_beta,
                                               hbuf, emb, caps, t, xbuf);
        // gates = [x_emb|z|h] @ [W_ih;W_hh] + (b_ih+b_hh)
        gemm(xbuf, WcombT, bias_comb, gatesbuf, B_, 4 * DEC_, XDIM, 4 * DEC_, nullptr, 0);
        // LSTM pointwise update
        lstm_update<<<(B_ * DEC_ + 255) / 256, 256, 0, stream>>>(gatesbuf, hbuf, cbuf,
                                                                 h_bf, hnew_bf, dlen, t);
        // preds[:, t, :] = mask ? h_new @ W_fc + b_fc : 0  (strided output rows)
        gemm(hnew_bf, WfcT, b_fc, out_preds + (size_t)t * VOCAB_,
             B_, VOCAB_, DEC_, T_ * VOCAB_, dlen, t);
    }
}
